// UBlock_55413668053594
// MI455X (gfx1250) — matrix operand order repaired and checked by reading, not compiled, not run on hardware
//
#include <hip/hip_runtime.h>
#include <hip/hip_bf16.h>
#include <math.h>

typedef __attribute__((ext_vector_type(16))) _Float16 v16h;
typedef __attribute__((ext_vector_type(8)))  _Float16 v8h;
typedef __attribute__((ext_vector_type(4)))  _Float16 v4h;
typedef __attribute__((ext_vector_type(8)))  float    v8f;

#define N_PTS 8192
#define S_NB  16
#define BN_INV_F 0.9999950000374998f   // 1/sqrt(1+1e-5)

// ---------------------------------------------------------------- WMMA helpers
__device__ __forceinline__ v8f wmma16(v16h a, v16h b, v8f c) {
  return __builtin_amdgcn_wmma_f32_16x16x32_f16(false, a, false, b, (short)0, c,
                                                false, false);
}

__device__ __forceinline__ v16h cat8(v8h lo, v8h hi) {
  return __builtin_shufflevector(lo, hi, 0, 1, 2, 3, 4, 5, 6, 7,
                                         8, 9, 10, 11, 12, 13, 14, 15);
}

// A fragment: 16x32 f16 tile, row-major LDS, 16B-aligned rows.
// lanes 0-15 (row=lane):   elems 0..15 = K 0..15
// lanes 16-31 (row=ln-16): elems 0..15 = K 16..31
// -> two contiguous 16B loads per lane.
__device__ __forceinline__ v16h load_a16(const _Float16* A, int lda, int lane) {
  int row = lane & 15;
  int kb  = (lane & 16) ? 16 : 0;
  const _Float16* base = A + row * lda + kb;
  v8h lo = *(const v8h*)(base);
  v8h hi = *(const v8h*)(base + 8);
  return cat8(lo, hi);
}

// B fragment from pre-swizzled LDS weights: [tile][lane][16 halves] contiguous.
__device__ __forceinline__ v16h load_b_swz(const _Float16* B, int tile, int lane) {
  const _Float16* base = B + tile * 512 + lane * 16;
  v8h lo = *(const v8h*)(base);
  v8h hi = *(const v8h*)(base + 8);
  return cat8(lo, hi);
}

// Stage a row-major KxN f32 weight into swizzled f16 fragments.
// tile = (k>>5)*NT + (n>>4); lane = (n&15) | (((k>>4)&1)<<4); e = k&15.
__device__ __forceinline__ void stage_b_swz4(_Float16* dst, const float* W,
                                             int Ktot, int Ntot, int NT, int tid) {
  const int total = (Ktot * Ntot) >> 2;
  for (int q = tid; q < total; q += 256) {
    int t = q << 2;
    int k = t / Ntot;                 // Ntot literal at call site -> shift
    int n = t - k * Ntot;
    const float4 w = *(const float4*)(W + t);
    int tile = (k >> 5) * NT + (n >> 4);
    int lane = (n & 15) | (((k >> 4) & 1) << 4);
    int e = k & 15;
    _Float16* p = dst + tile * 512 + lane * 16 + e;
    p[0]  = (_Float16)w.x;
    p[16] = (_Float16)w.y;
    p[32] = (_Float16)w.z;
    p[48] = (_Float16)w.w;
  }
}

// ---------------------------------------------------------------- KNN (exact)
__global__ void __launch_bounds__(256) knn_kernel(const int* __restrict__ idx,
                                                  int* __restrict__ knn) {
  __shared__ float sx[256], sy[256], sz[256], ss[256];
  const int i = blockIdx.x * 256 + threadIdx.x;
  const float xi = (float)idx[i * 4 + 1];
  const float yi = (float)idx[i * 4 + 2];
  const float zi = (float)idx[i * 4 + 3];
  const float sqi = xi * xi + yi * yi + zi * zi;
  float bd[S_NB];
  int   bi[S_NB];
#pragma unroll
  for (int k = 0; k < S_NB; ++k) { bd[k] = 3.0e38f; bi[k] = 0; }
  for (int t = 0; t < N_PTS; t += 256) {
    __syncthreads();
    int j = t + threadIdx.x;
    float xj = (float)idx[j * 4 + 1];
    float yj = (float)idx[j * 4 + 2];
    float zj = (float)idx[j * 4 + 3];
    sx[threadIdx.x] = xj; sy[threadIdx.x] = yj; sz[threadIdx.x] = zj;
    ss[threadIdx.x] = xj * xj + yj * yj + zj * zj;
    if (t + 256 < N_PTS)
      __builtin_prefetch(&idx[(t + 256 + (int)threadIdx.x) * 4], 0, 0);
    __syncthreads();
    for (int jj = 0; jj < 256; ++jj) {
      float d = sqi + ss[jj] - 2.0f * (xi * sx[jj] + yi * sy[jj] + zi * sz[jj]);
      if (d < bd[S_NB - 1]) {           // stable insertion: ties keep lower index
        float cd = d; int ci = t + jj;
#pragma unroll
        for (int k = 0; k < S_NB; ++k) {
          if (cd < bd[k]) {
            float td = bd[k]; int ti = bi[k];
            bd[k] = cd; bi[k] = ci; cd = td; ci = ti;
          }
        }
      }
    }
  }
#pragma unroll
  for (int k = 0; k < S_NB; ++k) knn[i * S_NB + k] = bi[k];
}

// ---------------------------------------------------------------- QKV GEMMs
__global__ void __launch_bounds__(256) qkv_kernel(
    const float* __restrict__ feats,
    const float* __restrict__ Wq, const float* __restrict__ bq,
    const float* __restrict__ Wk, const float* __restrict__ bk,
    const float* __restrict__ Wv, const float* __restrict__ bv,
    float* __restrict__ xq, float* __restrict__ xk, float* __restrict__ xv) {
  __shared__ __align__(16) _Float16 sWs[3][8 * 512];   // swizzled, KT=2 NT=4
  __shared__ float    sB[3][64];
  __shared__ __align__(16) _Float16 sA[8][16][64];
  const int tid = threadIdx.x;
  stage_b_swz4(sWs[0], Wq, 64, 64, 4, tid);
  stage_b_swz4(sWs[1], Wk, 64, 64, 4, tid);
  stage_b_swz4(sWs[2], Wv, 64, 64, 4, tid);
  if (tid < 64) { sB[0][tid] = bq[tid]; sB[1][tid] = bk[tid]; sB[2][tid] = bv[tid]; }
  const int wv = tid >> 5, lane = tid & 31;
  const int rowbase = blockIdx.x * 128 + wv * 16;
  for (int q = lane; q < 256; q += 32) {             // 16x64 tile as float4s
    int t = q << 2;
    int r = t >> 6, c = t & 63;
    const float4 f = *(const float4*)(feats + (rowbase + r) * 64 + c);
    v4h h; h[0] = (_Float16)f.x; h[1] = (_Float16)f.y;
           h[2] = (_Float16)f.z; h[3] = (_Float16)f.w;
    *(v4h*)(&sA[wv][r][c]) = h;
  }
  __syncthreads();
  for (int m = 0; m < 3; ++m) {
    float* out = (m == 0) ? xq : ((m == 1) ? xk : xv);
    for (int nt = 0; nt < 4; ++nt) {
      v8f acc = {};
#pragma unroll
      for (int k = 0; k < 2; ++k) {
        v16h a = load_a16(&sA[wv][0][k * 32], 64, lane);
        v16h b = load_b_swz(sWs[m], k * 4 + nt, lane);
        acc = wmma16(a, b, acc);
      }
      int ncol = nt * 16 + (lane & 15);
      float bias = sB[m][ncol];
#pragma unroll
      for (int r = 0; r < 8; ++r) {
        int mr = r + 8 * (lane >> 4);
        out[(rowbase + mr) * 64 + ncol] = acc[r] + bias;
      }
    }
  }
}

// ---------------------------------------------------------------- attention core
__global__ void __launch_bounds__(256) attn_kernel(
    const int* __restrict__ idx, const int* __restrict__ knn,
    const float* __restrict__ xq, const float* __restrict__ xk,
    const float* __restrict__ xv,
    const float* __restrict__ Wp1, const float* __restrict__ pg,
    const float* __restrict__ pb,  const float* __restrict__ Wp2,
    const float* __restrict__ bp2,
    const float* __restrict__ wg1, const float* __restrict__ wb1,
    const float* __restrict__ Ww1, const float* __restrict__ wg2,
    const float* __restrict__ wb2, const float* __restrict__ Ww2,
    const float* __restrict__ bw2, float* __restrict__ attn) {
  __shared__ __align__(16) _Float16 sWw1s[16 * 512];  // 16KB KT=2 NT=8
  __shared__ __align__(16) _Float16 sWw2s[16 * 512];  // 16KB KT=4 NT=4
  __shared__ __align__(16) _Float16 sWp2s[4 * 512];   // 4KB  KT=1 NT=4 (K pad 32)
  __shared__ __align__(16) _Float16 sApp[16][32];
  __shared__ __align__(16) _Float16 sAw[16][64];
  __shared__ __align__(16) _Float16 sW1[16][128];
  __shared__ __align__(16) float sPr[16][64];
  __shared__ __align__(16) float sVpr[16][64];
  __shared__ float sW2[16][64];
  __shared__ float sXq[64];
  __shared__ float sWp1[48], sPg[16], sPb[16], sBp2[64];
  __shared__ float sG1[64], sB1[64], sG2[128], sB2[128], sBw2[64];
  __shared__ float sXyz[3];
  __shared__ int   sKnn[16];
  const int tid = threadIdx.x, wv = tid >> 5, lane = tid & 31;

  stage_b_swz4(sWw1s, Ww1, 64, 128, 8, tid);
  stage_b_swz4(sWw2s, Ww2, 128, 64, 4, tid);
  for (int t = tid; t < 32 * 64; t += 256) {          // Wp2, K padded to 32
    int k = t >> 6, n = t & 63;
    float v = (k < 16) ? Wp2[k * 64 + n] : 0.f;
    int lane2 = (n & 15) | (((k >> 4) & 1) << 4);
    sWp2s[(n >> 4) * 512 + lane2 * 16 + (k & 15)] = (_Float16)v;
  }
  if (tid < 48) sWp1[tid] = Wp1[tid];
  if (tid < 16) { sPg[tid] = pg[tid]; sPb[tid] = pb[tid]; }
  if (tid < 64) { sBp2[tid] = bp2[tid]; sG1[tid] = wg1[tid];
                  sB1[tid] = wb1[tid]; sBw2[tid] = bw2[tid]; }
  if (tid < 128) { sG2[tid] = wg2[tid]; sB2[tid] = wb2[tid]; }

  for (int p = 0; p < 4; ++p) {
    const int n = blockIdx.x * 4 + p;
    __syncthreads();                    // weights ready / prev-point buffers free
    if (tid < 64) sXq[tid] = xq[n * 64 + tid];
    if (tid < 16) sKnn[tid] = knn[n * 16 + tid];
    if (tid < 3)  sXyz[tid] = (float)idx[n * 4 + 1 + tid];
    __syncthreads();
    // pos MLP hidden: pp = relu(bn(p_r @ Wp1))  (16x16, one element/thread)
    {
      int s = tid >> 4, h = tid & 15;
      int j = sKnn[s];
      float acc = 0.f;
#pragma unroll
      for (int d = 0; d < 3; ++d) {
        float pr = (float)idx[j * 4 + 1 + d] - sXyz[d];
        acc += pr * sWp1[d * 16 + h];
      }
      acc = acc * (sPg[h] * BN_INV_F) + sPb[h];
      sApp[s][h] = (_Float16)fmaxf(acc, 0.f);
      sApp[s][h + 16] = (_Float16)0.0f;
    }
    __syncthreads();
    // p_r = pp @ Wp2 + bp2 (16x64): waves 0-3, one WMMA each
    if (wv < 4) {
      v16h a = load_a16(&sApp[0][0], 32, lane);
      v16h b = load_b_swz(sWp2s, wv, lane);
      v8f acc = {};
      acc = wmma16(a, b, acc);
      int c = wv * 16 + (lane & 15);
#pragma unroll
      for (int r = 0; r < 8; ++r)
        sPr[r + 8 * (lane >> 4)][c] = acc[r] + sBp2[c];
    }
    __syncthreads();
    // gather K/V (float4 per row chunk); w0 = k_g - q + p_r -> bn relu f16
    {
      int s = tid >> 4, c0 = (tid & 15) << 2;        // 256 groups, one per thread
      int j = sKnn[s];
      const float4 kf = *(const float4*)(xk + j * 64 + c0);
      const float4 vf = *(const float4*)(xv + j * 64 + c0);
      const float4 pr = *(const float4*)(&sPr[s][c0]);
      const float* kp = (const float*)&kf;
      const float* vp = (const float*)&vf;
      const float* pp = (const float*)&pr;
      v4h aw; float4 vpr;
      float* vq = (float*)&vpr;
#pragma unroll
      for (int e = 0; e < 4; ++e) {
        int c = c0 + e;
        float w0 = kp[e] - sXq[c] + pp[e];
        w0 = w0 * (sG1[c] * BN_INV_F) + sB1[c];
        aw[e] = (_Float16)fmaxf(w0, 0.f);
        vq[e] = vp[e] + pp[e];
      }
      *(v4h*)(&sAw[s][c0]) = aw;
      *(float4*)(&sVpr[s][c0]) = vpr;
    }
    __syncthreads();
    // GEMM1: (16x64)@(64x128), 8 col-tiles, bn+relu -> sW1
    {
      v8f acc = {};
#pragma unroll
      for (int k = 0; k < 2; ++k) {
        v16h a = load_a16(&sAw[0][k * 32], 64, lane);
        v16h b = load_b_swz(sWw1s, k * 8 + wv, lane);
        acc = wmma16(a, b, acc);
      }
      int c = wv * 16 + (lane & 15);
      float g = sG2[c] * BN_INV_F, bb = sB2[c];
#pragma unroll
      for (int r = 0; r < 8; ++r)
        sW1[r + 8 * (lane >> 4)][c] = (_Float16)fmaxf(acc[r] * g + bb, 0.f);
    }
    __syncthreads();
    // GEMM2: (16x128)@(128x64)+bw2 -> sW2: waves 0-3, 4-WMMA K chain
    if (wv < 4) {
      v8f acc = {};
#pragma unroll
      for (int k = 0; k < 4; ++k) {
        v16h a = load_a16(&sW1[0][k * 32], 128, lane);
        v16h b = load_b_swz(sWw2s, k * 4 + wv, lane);
        acc = wmma16(a, b, acc);
      }
      int c = wv * 16 + (lane & 15);
#pragma unroll
      for (int r = 0; r < 8; ++r)
        sW2[r + 8 * (lane >> 4)][c] = acc[r] + sBw2[c];
    }
    __syncthreads();
    // softmax over S per channel + weighted sum with vpr
    if (tid < 64) {
      int c = tid;
      float m = -3.0e38f;
#pragma unroll
      for (int s = 0; s < 16; ++s) m = fmaxf(m, sW2[s][c]);
      float e[16], sum = 0.f;
#pragma unroll
      for (int s = 0; s < 16; ++s) { e[s] = expf(sW2[s][c] - m); sum += e[s]; }
      float a = 0.f;
#pragma unroll
      for (int s = 0; s < 16; ++s) a += sVpr[s][c] * e[s];
      attn[n * 64 + c] = a / sum;
    }
  }
}

// ---------------------------------------------------------------- head MLP + LN
__global__ void __launch_bounds__(256) head_kernel(
    const float* __restrict__ feats, const float* __restrict__ attn,
    const float* __restrict__ Wc1, const float* __restrict__ cg1,
    const float* __restrict__ cb1, const float* __restrict__ Wc2,
    const float* __restrict__ cg2, const float* __restrict__ cb2,
    const float* __restrict__ Wc3, const float* __restrict__ bc3,
    const float* __restrict__ lng, const float* __restrict__ lnb,
    float* __restrict__ out) {
  __shared__ __align__(16) _Float16 sW[32 * 512];     // 32KB swizzled weight buf
  __shared__ __align__(16) _Float16 sHa[32][128];
  __shared__ __align__(16) _Float16 sHb[32][128];
  __shared__ float    sOut[32][64];
  __shared__ float sG1[128], sB1[128], sG2[128], sB2[128], sBc3[64], sLg[64], sLb[64];
  const int tid = threadIdx.x, wv = tid >> 5, lane = tid & 31;
  const int rowbase = blockIdx.x * 32;
  if (tid < 128) { sG1[tid] = cg1[tid]; sB1[tid] = cb1[tid];
                   sG2[tid] = cg2[tid]; sB2[tid] = cb2[tid]; }
  if (tid < 64)  { sBc3[tid] = bc3[tid]; sLg[tid] = lng[tid]; sLb[tid] = lnb[tid]; }
  for (int q = tid; q < 1024; q += 256) {             // h = [feats||attn] as f16
    int t = q << 2;
    int r = t >> 7, c = t & 127;
    const float4 f = (c < 64) ? *(const float4*)(feats + (rowbase + r) * 64 + c)
                              : *(const float4*)(attn + (rowbase + r) * 64 + (c - 64));
    v4h h; h[0] = (_Float16)f.x; h[1] = (_Float16)f.y;
           h[2] = (_Float16)f.z; h[3] = (_Float16)f.w;
    *(v4h*)(&sHa[r][c]) = h;
  }
  stage_b_swz4(sW, Wc1, 128, 128, 8, tid);
  __syncthreads();
#pragma unroll
  for (int rt = 0; rt < 2; ++rt) {                    // stage 1 -> sHb
    v8f acc = {};
#pragma unroll
    for (int k = 0; k < 4; ++k) {
      v16h a = load_a16(&sHa[rt * 16][k * 32], 128, lane);
      v16h b = load_b_swz(sW, k * 8 + wv, lane);
      acc = wmma16(a, b, acc);
    }
    int c = wv * 16 + (lane & 15);
    float g = sG1[c] * BN_INV_F, bb = sB1[c];
#pragma unroll
    for (int r = 0; r < 8; ++r)
      sHb[rt * 16 + r + 8 * (lane >> 4)][c] = (_Float16)fmaxf(acc[r] * g + bb, 0.f);
  }
  __syncthreads();
  stage_b_swz4(sW, Wc2, 128, 128, 8, tid);
  __syncthreads();
#pragma unroll
  for (int rt = 0; rt < 2; ++rt) {                    // stage 2 -> sHa
    v8f acc = {};
#pragma unroll
    for (int k = 0; k < 4; ++k) {
      v16h a = load_a16(&sHb[rt * 16][k * 32], 128, lane);
      v16h b = load_b_swz(sW, k * 8 + wv, lane);
      acc = wmma16(a, b, acc);
    }
    int c = wv * 16 + (lane & 15);
    float g = sG2[c] * BN_INV_F, bb = sB2[c];
#pragma unroll
    for (int r = 0; r < 8; ++r)
      sHa[rt * 16 + r + 8 * (lane >> 4)][c] = (_Float16)fmaxf(acc[r] * g + bb, 0.f);
  }
  __syncthreads();
  stage_b_swz4(sW, Wc3, 128, 64, 4, tid);
  __syncthreads();
  {                                                   // stage 3: (32x128)@(128x64)
    int rt = wv >> 2, nt = wv & 3;
    v8f acc = {};
#pragma unroll
    for (int k = 0; k < 4; ++k) {
      v16h a = load_a16(&sHa[rt * 16][k * 32], 128, lane);
      v16h b = load_b_swz(sW, k * 4 + nt, lane);
      acc = wmma16(a, b, acc);
    }
    int c = nt * 16 + (lane & 15);
#pragma unroll
    for (int r = 0; r < 8; ++r)
      sOut[rt * 16 + r + 8 * (lane >> 4)][c] = acc[r] + sBc3[c];
  }
  __syncthreads();
  if (tid < 32) {                                     // layernorm over 64 channels
    int row = tid;
    float m = 0.f;
#pragma unroll
    for (int c = 0; c < 64; ++c) m += sOut[row][c];
    m *= (1.f / 64.f);
    float v = 0.f;
#pragma unroll
    for (int c = 0; c < 64; ++c) { float d = sOut[row][c] - m; v += d * d; }
    v *= (1.f / 64.f);
    float inv = 1.f / sqrtf(v + 1e-5f);
    for (int c = 0; c < 64; ++c)
      out[(rowbase + row) * 64 + c] = (sOut[row][c] - m) * inv * sLg[c] + sLb[c];
  }
}

// ---------------------------------------------------------------- launcher
extern "C" void kernel_launch(void* const* d_in, const int* in_sizes, int n_in,
                              void* d_out, int out_size, void* d_ws, size_t ws_size,
                              hipStream_t stream) {
  const int*   indices = (const int*)d_in[0];
  const float* feats = (const float*)d_in[1];
  const float* Wq  = (const float*)d_in[2];  const float* bq  = (const float*)d_in[3];
  const float* Wk  = (const float*)d_in[4];  const float* bk  = (const float*)d_in[5];
  const float* Wv  = (const float*)d_in[6];  const float* bv  = (const float*)d_in[7];
  const float* Wp1 = (const float*)d_in[8];  const float* pg  = (const float*)d_in[9];
  const float* pb  = (const float*)d_in[10]; const float* Wp2 = (const float*)d_in[11];
  const float* bp2 = (const float*)d_in[12];
  const float* wg1 = (const float*)d_in[13]; const float* wb1 = (const float*)d_in[14];
  const float* Ww1 = (const float*)d_in[15];
  const float* wg2 = (const float*)d_in[16]; const float* wb2 = (const float*)d_in[17];
  const float* Ww2 = (const float*)d_in[18]; const float* bw2 = (const float*)d_in[19];
  const float* Wc1 = (const float*)d_in[20]; const float* cg1 = (const float*)d_in[21];
  const float* cb1 = (const float*)d_in[22];
  const float* Wc2 = (const float*)d_in[23]; const float* cg2 = (const float*)d_in[24];
  const float* cb2 = (const float*)d_in[25];
  const float* Wc3 = (const float*)d_in[26]; const float* bc3 = (const float*)d_in[27];
  const float* lng = (const float*)d_in[28]; const float* lnb = (const float*)d_in[29];

  char* ws = (char*)d_ws;
  int*   knn  = (int*)ws;                                  // 8192*16*4  = 512KB
  float* xq   = (float*)(ws + 524288);                     // 2MB
  float* xk   = (float*)(ws + 524288 + 2097152);           // 2MB
  float* xv   = (float*)(ws + 524288 + 2 * 2097152);       // 2MB
  float* attn = (float*)(ws + 524288 + 3 * 2097152);       // 2MB

  knn_kernel <<<N_PTS / 256, 256, 0, stream>>>(indices, knn);
  qkv_kernel <<<N_PTS / 128, 256, 0, stream>>>(feats, Wq, bq, Wk, bk, Wv, bv,
                                               xq, xk, xv);
  attn_kernel<<<N_PTS / 4,   256, 0, stream>>>(indices, knn, xq, xk, xv,
                                               Wp1, pg, pb, Wp2, bp2,
                                               wg1, wb1, Ww1, wg2, wb2, Ww2, bw2,
                                               attn);
  head_kernel<<<N_PTS / 32,  256, 0, stream>>>(feats, attn, Wc1, cg1, cb1,
                                               Wc2, cg2, cb2, Wc3, bc3,
                                               lng, lnb, (float*)d_out);
}
